// HungarianMatcher_85916525789875
// MI455X (gfx1250) — matrix operand order repaired and checked by reading, not compiled, not run on hardware
//
#include <hip/hip_runtime.h>
#include <hip/hip_bf16.h>

typedef _Float16 v16h __attribute__((ext_vector_type(16)));
typedef float    v8f  __attribute__((ext_vector_type(8)));

#define BSZ   16
#define NQ    1000
#define NCLS  80
#define NT    100
#define NIGN  8
#define TPAD  112           // 7 tiles of 16 targets
#define QW    16            // queries per wave (WMMA M)
#define NWAVE 4
#define QB    (QW * NWAVE)  // 64 queries per block

union V16H { v16h v; _Float16 e[16]; };

__global__ __launch_bounds__(128)
void matcher_cost_kernel(const float* __restrict__ logits,   // (BS,Q,NC)
                         const float* __restrict__ pboxes,   // (BS,Q,4)
                         const float* __restrict__ tboxes,   // (BS*T,4)
                         const float* __restrict__ iboxes,   // (BS*NI,4)
                         const float* __restrict__ img,      // (BS,4)
                         const float* __restrict__ imgt,     // (BS*T,4)
                         const int*   __restrict__ tids,     // (BS*T,)
                         float* __restrict__ outC,           // (BS,Q,T)
                         float* __restrict__ outI)           // (BS,Q,NI)
{
    __shared__ float s_tb_raw[NT * 4];
    __shared__ float s_tb_nrm[NT * 4];
    __shared__ int   s_tid[TPAD];
    __shared__ float s_ign[NIGN * 4];
    __shared__ float s_qb [QB * 4];
    __shared__ float s_qbn[QB * 4];
    __shared__ float s_F[NWAVE][QW][NCLS];

    const int tx   = threadIdx.x;
    const int w    = tx >> 5;
    const int lane = tx & 31;
    const int bb   = blockIdx.y;
    const int qblk = blockIdx.x * QB;
    const int qbase = qblk + w * QW;

    // ---------------- load / precompute phase ----------------
    if (tx < TPAD) s_tid[tx] = (tx < NT) ? tids[bb * NT + tx] : -1;

    for (int i = tx; i < NT * 4; i += 128) {
        float v = tboxes[bb * NT * 4 + i];
        s_tb_raw[i] = v;
        s_tb_nrm[i] = v / imgt[bb * NT * 4 + i];
    }
    if (tx < NIGN * 4) s_ign[tx] = iboxes[bb * NIGN * 4 + tx];

    for (int i = tx; i < QB * 4; i += 128) {
        int qloc = i >> 2, k = i & 3;
        int q = qblk + qloc; if (q > NQ - 1) q = NQ - 1;
        float v = pboxes[(bb * NQ + q) * 4 + k];
        s_qb[i]  = v;
        s_qbn[i] = v / img[bb * 4 + k];
    }

    // focal classification cost matrix F (16 x 80) per wave
    for (int i = lane; i < QW * NCLS; i += 32) {
        int qloc = i / NCLS, c = i - qloc * NCLS;
        int q = qbase + qloc; if (q > NQ - 1) q = NQ - 1;
        float x = logits[(bb * NQ + q) * NCLS + c];
        float p = 1.0f / (1.0f + __expf(-x));
        float pos = 0.25f * (1.0f - p) * (1.0f - p) * (-__logf(p + 1e-8f));
        float neg = 0.75f * p * p * (-__logf(1.0f - p + 1e-8f));
        s_F[w][qloc][c] = pos - neg;
    }
    __syncthreads();

    // ---------------- compute phase (per wave: 16 queries x 100 targets) ----
    const int m  = lane & 15;   // M-row held by this lane in the A operand
    const int kh = lane >> 4;   // lane half (selects K sub-range)

    // Preload A operand: focal matrix split into f16 hi + f16 residual.
    // 16-bit A 16x32 layout: half h<8 -> K = 8*kh + h ; h>=8 -> K = 16 + 8*kh + (h-8)
    V16H ahi[3], alo[3];
#pragma unroll
    for (int kk = 0; kk < 3; ++kk) {
#pragma unroll
        for (int h = 0; h < 16; ++h) {
            int K = (h < 8) ? (8 * kh + h) : (16 + 8 * kh + (h - 8));
            int c = 32 * kk + K;
            float f = (c < NCLS) ? s_F[w][m][c] : 0.0f;
            _Float16 hi = (_Float16)f;
            ahi[kk].e[h] = hi;
            alo[kk].e[h] = (_Float16)(f - (float)hi);
        }
    }

    const int nrel = lane & 15;  // N-column held by this lane in B/D operands
#pragma unroll 1
    for (int j = 0; j < 7; ++j) {
        int t = 16 * j + nrel;
        int mytid = s_tid[t];    // -1 for padded targets -> one-hot all zero

        v8f acc = {};
#pragma unroll
        for (int kk = 0; kk < 3; ++kk) {
            // B 32x16 layout: half h<8 -> K = 8*kh + h ; h>=8 -> K = 16 + 8*kh + (h-8); one-hot of mytid
            V16H bu;
#pragma unroll
            for (int h = 0; h < 16; ++h)
                bu.e[h] = (mytid == (32 * kk + ((h < 8) ? (8 * kh + h) : (16 + 8 * kh + (h - 8))))) ? (_Float16)1.0f
                                                             : (_Float16)0.0f;
            acc = __builtin_amdgcn_wmma_f32_16x16x32_f16(
                      false, ahi[kk].v, false, bu.v, (short)0, acc, false, false);
            acc = __builtin_amdgcn_wmma_f32_16x16x32_f16(
                      false, alo[kk].v, false, bu.v, (short)0, acc, false, false);
        }

        // per-lane target data (clamp LDS index for padded lanes; store is guarded)
        int tl = (t < NT) ? t : (NT - 1);
        float tx1 = s_tb_raw[4 * tl + 0], ty1 = s_tb_raw[4 * tl + 1];
        float tx2 = s_tb_raw[4 * tl + 2], ty2 = s_tb_raw[4 * tl + 3];
        float tn0 = s_tb_nrm[4 * tl + 0], tn1 = s_tb_nrm[4 * tl + 1];
        float tn2 = s_tb_nrm[4 * tl + 2], tn3 = s_tb_nrm[4 * tl + 3];
        float ta  = (tx2 - tx1) * (ty2 - ty1);

#pragma unroll
        for (int r = 0; r < 8; ++r) {
            int qloc = r + 8 * kh;                 // D layout: M = r + 8*(lane>=16)
            int qi = (16 * w + qloc) * 4;
            float px1 = s_qb[qi + 0], py1 = s_qb[qi + 1];
            float px2 = s_qb[qi + 2], py2 = s_qb[qi + 3];
            float pa = (px2 - px1) * (py2 - py1);

            float iw = fmaxf(fminf(px2, tx2) - fmaxf(px1, tx1), 0.0f);
            float ih = fmaxf(fminf(py2, ty2) - fmaxf(py1, ty1), 0.0f);
            float inter = iw * ih;
            float uni = pa + ta - inter;
            float iou = inter / uni;

            float ew = fmaxf(fmaxf(px2, tx2) - fminf(px1, tx1), 0.0f);
            float eh = fmaxf(fmaxf(py2, ty2) - fminf(py1, ty1), 0.0f);
            float enc = ew * eh;
            float giou = iou - (enc - uni) / enc;

            float cbb = fabsf(s_qbn[qi + 0] - tn0) + fabsf(s_qbn[qi + 1] - tn1) +
                        fabsf(s_qbn[qi + 2] - tn2) + fabsf(s_qbn[qi + 3] - tn3);

            float cost = 5.0f * cbb + 2.0f * acc[r] - 2.0f * giou;

            int q = qbase + qloc;
            if (q < NQ && t < NT)
                outC[(bb * NQ + q) * NT + t] = cost;
        }
    }

    // ---------------- ignore-box IOA epilogue ----------------
    for (int idx = lane; idx < QW * NIGN; idx += 32) {
        int qloc = idx >> 3, ig = idx & 7;
        int qi = (16 * w + qloc) * 4;
        float px1 = s_qb[qi + 0], py1 = s_qb[qi + 1];
        float px2 = s_qb[qi + 2], py2 = s_qb[qi + 3];
        float pa = (px2 - px1) * (py2 - py1);
        float gx1 = s_ign[4 * ig + 0], gy1 = s_ign[4 * ig + 1];
        float gx2 = s_ign[4 * ig + 2], gy2 = s_ign[4 * ig + 3];
        float iw = fmaxf(fminf(px2, gx2) - fmaxf(px1, gx1), 0.0f);
        float ih = fmaxf(fminf(py2, gy2) - fmaxf(py1, gy1), 0.0f);
        int q = qbase + qloc;
        if (q < NQ)
            outI[(bb * NQ + q) * NIGN + ig] = (iw * ih) / pa;
    }
}

extern "C" void kernel_launch(void* const* d_in, const int* in_sizes, int n_in,
                              void* d_out, int out_size, void* d_ws, size_t ws_size,
                              hipStream_t stream) {
    (void)in_sizes; (void)n_in; (void)out_size; (void)d_ws; (void)ws_size;
    const float* logits = (const float*)d_in[0];
    const float* pboxes = (const float*)d_in[1];
    const float* tboxes = (const float*)d_in[2];
    const float* iboxes = (const float*)d_in[3];
    const float* img    = (const float*)d_in[4];
    const float* imgt   = (const float*)d_in[5];
    const int*   tids   = (const int*)  d_in[6];

    float* outC = (float*)d_out;
    float* outI = outC + (size_t)BSZ * NQ * NT;

    dim3 grid((NQ + QB - 1) / QB, BSZ);   // (16, 16)
    matcher_cost_kernel<<<grid, 128, 0, stream>>>(logits, pboxes, tboxes, iboxes,
                                                  img, imgt, tids, outC, outI);
}
